// NeoGNN_66992899883202
// MI455X (gfx1250) — matrix operand order repaired and checked by reading, not compiled, not run on hardware
//
#include <hip/hip_runtime.h>

// NeoGNN forward for MI455X (gfx1250, wave32, WMMA).
// All dense 128x128 GEMMs run through v_wmma_f32_16x16x32_f16 (f16 A/B staged
// in LDS, f32 accumulate). Each block computes a 32x128 output stripe with 8
// waves; each wave owns one 16-wide column tile and two 16-row tiles sharing
// the B fragment (3 LDS loads per WMMA). Edge message passing uses
// wave-per-edge float4 scatter with global f32 atomics; GAT softmax max uses a
// monotone-uint atomicMax mapping.

typedef __attribute__((ext_vector_type(16))) _Float16 v16h;
typedef __attribute__((ext_vector_type(8)))  float    v8f;

#define F   128
#define TPB 256

__device__ __forceinline__ float lrelu(float x) { return x > 0.f ? x : 0.2f * x; }
// monotone map float -> uint so unsigned atomicMax == float max
__device__ __forceinline__ unsigned fmapu(float f) {
  unsigned u = __float_as_uint(f);
  return (u & 0x80000000u) ? ~u : (u | 0x80000000u);
}
__device__ __forceinline__ float funmap(unsigned u) {
  return __uint_as_float((u & 0x80000000u) ? (u & 0x7fffffffu) : ~u);
}

// ---------------- elementwise helpers ----------------
__global__ void k_zero_f(float* p, int n) {
  int i = blockIdx.x * blockDim.x + threadIdx.x;
  if (i < n) p[i] = 0.f;
}
__global__ void k_f32_to_f16(const float* __restrict__ s, _Float16* __restrict__ d, int n) {
  int i = blockIdx.x * blockDim.x + threadIdx.x;
  if (i < n) d[i] = (_Float16)s[i];
}
// 128x128 f32 row-major [k][n] -> f16 column-major [n][k] (B-fragment friendly)
__global__ void k_wtrans(const float* __restrict__ W, _Float16* __restrict__ Wt) {
  int i = blockIdx.x * blockDim.x + threadIdx.x;
  if (i >= F * F) return;
  int n = i >> 7, k = i & 127;
  Wt[n * F + k] = (_Float16)W[k * F + n];
}

// ---------------- degrees ----------------
__global__ void k_degrees(const int* __restrict__ row, const int* __restrict__ col,
                          float* deg_in, float* deg_out, int E) {
  int e = blockIdx.x * blockDim.x + threadIdx.x;
  if (e < E) { atomicAdd(&deg_in[col[e]], 1.f); atomicAdd(&deg_out[row[e]], 1.f); }
}
__global__ void k_dinv(const float* __restrict__ deg_in, float* dinv, float* invdeg, int N) {
  int i = blockIdx.x * blockDim.x + threadIdx.x;
  if (i < N) {
    float d = deg_in[i];
    dinv[i]   = rsqrtf(d + 1.f);      // self-loop included (deg >= 1)
    invdeg[i] = 1.f / fmaxf(d, 1.f);  // SAGE mean denominator (no self-loop)
  }
}

// ---------------- WMMA GEMM: OUT[M x 128] (+)= act(A16 @ W + bias) ----------------
// A16: f16 row-major [M][128]; Wt: f16 column-major [n][k] (pre-repacked).
// Block = 8 waves = 32 rows x 128 cols. Per wave: 2 row tiles x 1 col tile,
// B fragment shared across the two WMMAs of each K-step. K=128 -> 8 wmma/wave.
template <int BIAS, int RELU, int ACCUM>
__global__ void __launch_bounds__(TPB)
k_gemm128(const _Float16* __restrict__ A, const _Float16* __restrict__ Wt,
          const float* __restrict__ bias, float* __restrict__ OUT, int M) {
  __shared__ _Float16 sW[F * F];   // 32 KB, column-major [n][k]
  __shared__ _Float16 sA[32 * F];  // 8 KB
  const int tid = threadIdx.x;
  {  // stage full weight (vectorized, 8 uint4 per thread)
    const uint4* src = (const uint4*)Wt;
    uint4* dst = (uint4*)sW;
#pragma unroll
    for (int i = 0; i < 8; ++i) dst[tid + TPB * i] = src[tid + TPB * i];
  }
  {  // stage 32 rows of A, zero-pad past M (512 uint4, 2 per thread)
#pragma unroll
    for (int i = 0; i < 2; ++i) {
      int idx = tid + TPB * i;
      int r = idx >> 4, q = idx & 15;
      int rg = blockIdx.x * 32 + r;
      uint4 v = make_uint4(0u, 0u, 0u, 0u);
      if (rg < M) v = ((const uint4*)(A + (size_t)rg * F))[q];
      ((uint4*)sA)[idx] = v;
    }
  }
  __syncthreads();

  const int lane = tid & 31;
  const int wv   = tid >> 5;       // 8 waves -> 8 column tiles
  const int half = lane >> 4;
  const int l16  = lane & 15;
  const int n    = wv * 16 + l16;  // global column 0..127

  v8f c0 = {0.f, 0.f, 0.f, 0.f, 0.f, 0.f, 0.f, 0.f};
  v8f c1 = {0.f, 0.f, 0.f, 0.f, 0.f, 0.f, 0.f, 0.f};
#pragma unroll
  for (int k0 = 0; k0 < F; k0 += 32) {
    union { v16h v; uint4 q[2]; } a0, a1, b;
    // B frag: column n, contiguous 16-K run per lane half (shared by both tiles)
    const int kb = k0 + half * 16;
    b.q[0] = *(const uint4*)&sW[n * F + kb];
    b.q[1] = *(const uint4*)&sW[n * F + kb + 8];
    // A frags: contiguous 16-K run per lane half (matches B): lane<16 -> K k0..k0+15; lane>=16 +16
    const int ka = k0 + half * 16;
    a0.q[0] = *(const uint4*)&sA[l16 * F + ka];
    a0.q[1] = *(const uint4*)&sA[l16 * F + ka + 8];
    a1.q[0] = *(const uint4*)&sA[(16 + l16) * F + ka];
    a1.q[1] = *(const uint4*)&sA[(16 + l16) * F + ka + 8];
    c0 = __builtin_amdgcn_wmma_f32_16x16x32_f16(false, a0.v, false, b.v, (short)0, c0, false, false);
    c1 = __builtin_amdgcn_wmma_f32_16x16x32_f16(false, a1.v, false, b.v, (short)0, c1, false, false);
  }

  // C layout: lanes->N, vgpr v -> M = 8*half+v
  const int rbase = blockIdx.x * 32 + half * 8;
  const float bv = BIAS ? bias[n] : 0.f;
  float* o0 = OUT + (size_t)rbase * F + n;
  if (rbase + 24 <= M) {  // fast path: both row tiles fully in range
#pragma unroll
    for (int v = 0; v < 8; ++v) {
      float v0 = c0[v] + bv, v1 = c1[v] + bv;
      if (RELU) { v0 = fmaxf(v0, 0.f); v1 = fmaxf(v1, 0.f); }
      float* p0 = o0 + (size_t)v * F;
      float* p1 = p0 + (size_t)16 * F;
      if (ACCUM) { *p0 += v0; *p1 += v1; } else { *p0 = v0; *p1 = v1; }
    }
  } else {
#pragma unroll
    for (int v = 0; v < 8; ++v) {
      float v0 = c0[v] + bv, v1 = c1[v] + bv;
      if (RELU) { v0 = fmaxf(v0, 0.f); v1 = fmaxf(v1, 0.f); }
      if (rbase + v < M) {
        float* p0 = o0 + (size_t)v * F;
        if (ACCUM) *p0 += v0; else *p0 = v0;
      }
      if (rbase + 16 + v < M) {
        float* p1 = o0 + (size_t)(16 + v) * F;
        if (ACCUM) *p1 += v1; else *p1 = v1;
      }
    }
  }
}

// ---------------- edge scatters (wave per edge, float4 per lane) ----------------
__global__ void k_scat(const int* __restrict__ ai, const int* __restrict__ bi,
                       const float* __restrict__ SRC, float* __restrict__ DST, int E) {
  int g = blockIdx.x * blockDim.x + threadIdx.x;
  int e = g >> 5, lane = g & 31;
  if (e >= E) return;
  int s = ai[e], d = bi[e];
  float4 v = ((const float4*)(SRC + (size_t)s * F))[lane];
  float* o = DST + (size_t)d * F + lane * 4;
  atomicAdd(o + 0, v.x); atomicAdd(o + 1, v.y);
  atomicAdd(o + 2, v.z); atomicAdd(o + 3, v.w);
}
__global__ void k_gcn_edge(const int* __restrict__ row, const int* __restrict__ col,
                           const float* __restrict__ XW, const float* __restrict__ dinv,
                           float* __restrict__ ACC, int E) {
  int g = blockIdx.x * blockDim.x + threadIdx.x;
  int e = g >> 5, lane = g & 31;
  if (e >= E) return;
  int s = row[e], d = col[e];
  float nrm = dinv[s] * dinv[d];
  float4 v = ((const float4*)(XW + (size_t)s * F))[lane];
  float* o = ACC + (size_t)d * F + lane * 4;
  atomicAdd(o + 0, nrm * v.x); atomicAdd(o + 1, nrm * v.y);
  atomicAdd(o + 2, nrm * v.z); atomicAdd(o + 3, nrm * v.w);
}

// gcn self-loop + bias, SAGE mean->f16, GIN (x+nsum)->f16, fused per element
__global__ void k_node_fused(const float* __restrict__ Xin, const float* __restrict__ XW,
                             const float* __restrict__ NSUM, const float* __restrict__ dinv,
                             const float* __restrict__ invdeg, const float* __restrict__ gcn_b,
                             float* __restrict__ ACC, _Float16* __restrict__ MEANh,
                             _Float16* __restrict__ GINh, int total) {
  int idx = blockIdx.x * blockDim.x + threadIdx.x;
  if (idx >= total) return;
  int i = idx >> 7, c = idx & 127;
  float ns = NSUM[idx], dv = dinv[i];
  ACC[idx] += XW[idx] * dv * dv + gcn_b[c];
  MEANh[idx] = (_Float16)(ns * invdeg[i]);
  GINh[idx]  = (_Float16)(Xin[idx] + ns);
}

// ---------------- GAT ----------------
__global__ void k_gemv2(const float* __restrict__ XG, const float* __restrict__ asrc,
                        const float* __restrict__ adst, float* a_s, float* a_d, int N) {
  int g = blockIdx.x * blockDim.x + threadIdx.x;
  int i = g >> 5, lane = g & 31;
  if (i >= N) return;
  float4 x = ((const float4*)(XG + (size_t)i * F))[lane];
  float4 s4 = ((const float4*)asrc)[lane];
  float4 d4 = ((const float4*)adst)[lane];
  float s = x.x * s4.x + x.y * s4.y + x.z * s4.z + x.w * s4.w;
  float d = x.x * d4.x + x.y * d4.y + x.z * d4.z + x.w * d4.w;
  for (int o = 16; o; o >>= 1) { s += __shfl_xor(s, o); d += __shfl_xor(d, o); }
  if (!lane) { a_s[i] = s; a_d[i] = d; }
}
__global__ void k_gat_minit(const float* a_s, const float* a_d, unsigned* mu, int N) {
  int i = blockIdx.x * blockDim.x + threadIdx.x;
  if (i < N) mu[i] = fmapu(lrelu(a_s[i] + a_d[i]));  // self-loop term seeds max
}
__global__ void k_gat_emax(const int* __restrict__ row, const int* __restrict__ col,
                           const float* a_s, const float* a_d, unsigned* mu, int E) {
  int e = blockIdx.x * blockDim.x + threadIdx.x;
  if (e < E) atomicMax(&mu[col[e]], fmapu(lrelu(a_s[row[e]] + a_d[col[e]])));
}
__global__ void k_gat_den0(const float* a_s, const float* a_d, const unsigned* mu,
                           float* m, float* den, int N) {
  int i = blockIdx.x * blockDim.x + threadIdx.x;
  if (i >= N) return;
  float mm = funmap(mu[i]);
  m[i] = mm;
  den[i] = expf(lrelu(a_s[i] + a_d[i]) - mm);  // self-loop contribution
}
__global__ void k_gat_eden(const int* __restrict__ row, const int* __restrict__ col,
                           const float* a_s, const float* a_d, const float* m,
                           float* den, int E) {
  int e = blockIdx.x * blockDim.x + threadIdx.x;
  if (e >= E) return;
  int r = row[e], c = col[e];
  atomicAdd(&den[c], expf(lrelu(a_s[r] + a_d[c]) - m[c]));
}
__global__ void k_gat_escat(const int* __restrict__ row, const int* __restrict__ col,
                            const float* a_s, const float* a_d, const float* m,
                            const float* den, const float* __restrict__ XG,
                            float* __restrict__ ACC, int E) {
  int g = blockIdx.x * blockDim.x + threadIdx.x;
  int e = g >> 5, lane = g & 31;
  if (e >= E) return;
  int s = row[e], d = col[e];
  float w = expf(lrelu(a_s[s] + a_d[d]) - m[d]) / den[d];
  float4 v = ((const float4*)(XG + (size_t)s * F))[lane];
  float* o = ACC + (size_t)d * F + lane * 4;
  atomicAdd(o + 0, w * v.x); atomicAdd(o + 1, w * v.y);
  atomicAdd(o + 2, w * v.z); atomicAdd(o + 3, w * v.w);
}
__global__ void k_gat_self_relu(const float* a_s, const float* a_d, const float* m,
                                const float* den, const float* __restrict__ XG,
                                const float* __restrict__ gat_b,
                                const float* __restrict__ ACC, float* __restrict__ Xout,
                                int total) {
  int idx = blockIdx.x * blockDim.x + threadIdx.x;
  if (idx >= total) return;
  int i = idx >> 7, c = idx & 127;
  float w = expf(lrelu(a_s[i] + a_d[i]) - m[i]) / den[i];
  Xout[idx] = fmaxf(ACC[idx] + w * XG[idx] + gat_b[c], 0.f);
}

// ---------------- decoder ----------------
__global__ void k_nmean_div(float* __restrict__ NM, const float* __restrict__ deg_out, int total) {
  int idx = blockIdx.x * blockDim.x + threadIdx.x;
  if (idx < total) NM[idx] *= 1.f / fmaxf(deg_out[idx >> 7], 1.f);
}
__global__ void k_neo(const int* __restrict__ src, const int* __restrict__ dst,
                      const float* __restrict__ NM, const float* __restrict__ neoW,
                      const float* __restrict__ neob, float* __restrict__ neo, int P) {
  int g = blockIdx.x * blockDim.x + threadIdx.x;
  int p = g >> 5, lane = g & 31;
  if (p >= P) return;
  float4 u = ((const float4*)(NM + (size_t)src[p] * F))[lane];
  float4 v = ((const float4*)(NM + (size_t)dst[p] * F))[lane];
  float4 w = ((const float4*)neoW)[lane];
  float acc = u.x * v.x * w.x + u.y * v.y * w.y + u.z * v.z * w.z + u.w * v.w * w.w;
  for (int o = 16; o; o >>= 1) acc += __shfl_xor(acc, o);
  if (!lane) neo[p] = 1.f / (1.f + expf(-(acc + neob[0])));
}
__global__ void k_gather(const int* __restrict__ src, const int* __restrict__ dst,
                         const float* __restrict__ X, _Float16* __restrict__ HU,
                         _Float16* __restrict__ HV, int total) {
  int idx = blockIdx.x * blockDim.x + threadIdx.x;
  if (idx >= total) return;
  int p = idx >> 7, c = idx & 127;
  HU[idx] = (_Float16)X[(size_t)src[p] * F + c];
  HV[idx] = (_Float16)X[(size_t)dst[p] * F + c];
}
// h1 = relu(T1 + neo*w1_last + b1); out = h1 @ dec_W2 + b2  (wave per pair)
__global__ void k_dec_final(const float* __restrict__ T1, const float* __restrict__ neo,
                            const float* __restrict__ w1l, const float* __restrict__ b1,
                            const float* __restrict__ W2, const float* __restrict__ b2,
                            float* __restrict__ out, int P) {
  int g = blockIdx.x * blockDim.x + threadIdx.x;
  int p = g >> 5, lane = g & 31;
  if (p >= P) return;
  float np = neo[p];
  float4 t  = ((const float4*)(T1 + (size_t)p * F))[lane];
  float4 wl = ((const float4*)w1l)[lane];
  float4 bb = ((const float4*)b1)[lane];
  float h0 = fmaxf(t.x + np * wl.x + bb.x, 0.f);
  float h1 = fmaxf(t.y + np * wl.y + bb.y, 0.f);
  float h2 = fmaxf(t.z + np * wl.z + bb.z, 0.f);
  float h3 = fmaxf(t.w + np * wl.w + bb.w, 0.f);
  float4 wa = ((const float4*)W2)[lane * 2];      // {W2[c0][0],W2[c0][1],W2[c0+1][0],W2[c0+1][1]}
  float4 wb = ((const float4*)W2)[lane * 2 + 1];
  float o0 = h0 * wa.x + h1 * wa.z + h2 * wb.x + h3 * wb.z;
  float o1 = h0 * wa.y + h1 * wa.w + h2 * wb.y + h3 * wb.w;
  for (int o = 16; o; o >>= 1) { o0 += __shfl_xor(o0, o); o1 += __shfl_xor(o1, o); }
  if (!lane) { out[p * 2 + 0] = o0 + b2[0]; out[p * 2 + 1] = o1 + b2[1]; }
}

// ---------------- host launcher ----------------
extern "C" void kernel_launch(void* const* d_in, const int* in_sizes, int n_in,
                              void* d_out, int out_size, void* d_ws, size_t ws_size,
                              hipStream_t stream) {
  (void)n_in; (void)out_size; (void)ws_size;
  const float* x   = (const float*)d_in[0];
  const int*   ei  = (const int*)d_in[1];
  const int*   eli = (const int*)d_in[2];
  const int N = in_sizes[0] / F;
  const int E = in_sizes[1] / 2;
  const int P = in_sizes[2] / 2;
  const int* row = ei,  *col  = ei  + E;
  const int* psrc = eli, *pdst = eli + P;

  // params flattened as jax pytree leaves (dict keys sorted): dec_W1, dec_b1,
  // dec_W2, dec_b2, layers[0..2]{gat_W,gat_adst,gat_asrc,gat_b,gcn_W,gcn_b,
  // gin_W1,gin_W2,gin_b1,gin_b2,sage_Wl,sage_Wr,sage_bl}, neo_W, neo_b.
  auto gp = [&](int i) { return (const float*)d_in[i]; };
  const float* dec_W1 = gp(3);
  const float* dec_b1 = gp(4);
  const float* dec_W2 = gp(5);
  const float* dec_b2 = gp(6);
  struct LP { const float *gat_W,*gat_adst,*gat_asrc,*gat_b,*gcn_W,*gcn_b,
              *gin_W1,*gin_W2,*gin_b1,*gin_b2,*sage_Wl,*sage_Wr,*sage_bl; } L[3];
  for (int l = 0; l < 3; ++l) {
    int b = 7 + 13 * l;
    L[l] = { gp(b+0), gp(b+1), gp(b+2), gp(b+3), gp(b+4), gp(b+5), gp(b+6),
             gp(b+7), gp(b+8), gp(b+9), gp(b+10), gp(b+11), gp(b+12) };
  }
  const float* neo_W = gp(46);
  const float* neo_b = gp(47);

  // ---- workspace carve-out (256B aligned) ----
  char* base = (char*)d_ws; size_t off = 0;
  auto WS = [&](size_t bytes) -> void* {
    void* p = base + off; off += (bytes + 255) & ~(size_t)255; return p;
  };
  float* X    = (float*)WS((size_t)N * F * 4);
  float* ACC  = (float*)WS((size_t)N * F * 4);
  float* TMP  = (float*)WS((size_t)N * F * 4);
  float* NSUM = (float*)WS((size_t)N * F * 4);
  float* XG   = (float*)WS((size_t)N * F * 4);
  _Float16* Xh    = (_Float16*)WS((size_t)N * F * 2);
  _Float16* MEANh = (_Float16*)WS((size_t)N * F * 2);  // reused as gin-mid f16
  _Float16* GINh  = (_Float16*)WS((size_t)N * F * 2);
  float* deg_in  = (float*)WS((size_t)N * 4);
  float* deg_out = (float*)WS((size_t)N * 4);
  float* dinv    = (float*)WS((size_t)N * 4);
  float* invdeg  = (float*)WS((size_t)N * 4);
  float* a_s     = (float*)WS((size_t)N * 4);
  float* a_d     = (float*)WS((size_t)N * 4);
  unsigned* mu   = (unsigned*)WS((size_t)N * 4);
  float* mf      = (float*)WS((size_t)N * 4);
  float* den     = (float*)WS((size_t)N * 4);
  float* neo     = (float*)WS((size_t)P * 4);
  _Float16* HU   = (_Float16*)WS((size_t)P * F * 2);
  _Float16* HV   = (_Float16*)WS((size_t)P * F * 2);
  float* T1p     = (float*)WS((size_t)P * F * 4);
  _Float16* Wt[3][6];  // gcn_W, sage_Wl, sage_Wr, gin_W1, gin_W2, gat_W
  for (int l = 0; l < 3; ++l)
    for (int j = 0; j < 6; ++j) Wt[l][j] = (_Float16*)WS((size_t)F * F * 2);
  _Float16* decA = (_Float16*)WS((size_t)F * F * 2);
  _Float16* decB = (_Float16*)WS((size_t)F * F * 2);

  const int NF = N * F, PF = P * F;
  dim3 blk(TPB);
  auto g1 = [&](int n) { return dim3((unsigned)((n + TPB - 1) / TPB)); };
  const dim3 gW(64);                           // 16384 weight elements
  const dim3 gemmN((unsigned)((N + 31) / 32)); // 32-row stripes
  const dim3 gemmP((unsigned)((P + 31) / 32));
  const dim3 edgeW(g1(E * 32));                // wave per edge
  const dim3 nodeW(g1(N * 32));                // wave per node
  const dim3 pairW(g1(P * 32));                // wave per pair

  // one-time weight repack f32 -> f16 column-major
  for (int l = 0; l < 3; ++l) {
    const float* src[6] = { L[l].gcn_W, L[l].sage_Wl, L[l].sage_Wr,
                            L[l].gin_W1, L[l].gin_W2, L[l].gat_W };
    for (int j = 0; j < 6; ++j) k_wtrans<<<gW, blk, 0, stream>>>(src[j], Wt[l][j]);
  }
  k_wtrans<<<gW, blk, 0, stream>>>(dec_W1,          decA);  // dec_W1 rows   0..127
  k_wtrans<<<gW, blk, 0, stream>>>(dec_W1 + F * F,  decB);  // dec_W1 rows 128..255

  // degrees (edge-structure only; shared by all layers)
  k_zero_f<<<g1(N), blk, 0, stream>>>(deg_in, N);
  k_zero_f<<<g1(N), blk, 0, stream>>>(deg_out, N);
  k_degrees<<<g1(E), blk, 0, stream>>>(row, col, deg_in, deg_out, E);
  k_dinv<<<g1(N), blk, 0, stream>>>(deg_in, dinv, invdeg, N);

  const float* Xin = x;
  for (int l = 0; l < 3; ++l) {
    const LP& p = L[l];
    k_f32_to_f16<<<g1(NF), blk, 0, stream>>>(Xin, Xh, NF);
    k_gemm128<0,0,0><<<gemmN, blk, 0, stream>>>(Xh, Wt[l][0], nullptr, TMP, N);  // xw = x@gcn_W
    k_zero_f<<<g1(NF), blk, 0, stream>>>(ACC, NF);
    k_zero_f<<<g1(NF), blk, 0, stream>>>(NSUM, NF);
    k_gcn_edge<<<edgeW, blk, 0, stream>>>(row, col, TMP, dinv, ACC, E);
    k_scat<<<edgeW, blk, 0, stream>>>(row, col, Xin, NSUM, E);                   // nsum
    k_node_fused<<<g1(NF), blk, 0, stream>>>(Xin, TMP, NSUM, dinv, invdeg,
                                             p.gcn_b, ACC, MEANh, GINh, NF);
    k_gemm128<1,0,1><<<gemmN, blk, 0, stream>>>(MEANh, Wt[l][1], p.sage_bl, ACC, N); // += mean@Wl+bl
    k_gemm128<0,0,1><<<gemmN, blk, 0, stream>>>(Xh,    Wt[l][2], nullptr,   ACC, N); // += x@Wr
    k_gemm128<1,1,0><<<gemmN, blk, 0, stream>>>(GINh,  Wt[l][3], p.gin_b1,  TMP, N); // relu(gin@W1+b1)
    k_f32_to_f16<<<g1(NF), blk, 0, stream>>>(TMP, MEANh, NF);
    k_gemm128<1,0,1><<<gemmN, blk, 0, stream>>>(MEANh, Wt[l][4], p.gin_b2,  ACC, N); // += gin2+b2
    k_gemm128<0,0,0><<<gemmN, blk, 0, stream>>>(Xh,    Wt[l][5], nullptr,   XG,  N); // xg = x@gat_W
    k_gemv2<<<nodeW, blk, 0, stream>>>(XG, p.gat_asrc, p.gat_adst, a_s, a_d, N);
    k_gat_minit<<<g1(N), blk, 0, stream>>>(a_s, a_d, mu, N);
    k_gat_emax<<<g1(E), blk, 0, stream>>>(row, col, a_s, a_d, mu, E);
    k_gat_den0<<<g1(N), blk, 0, stream>>>(a_s, a_d, mu, mf, den, N);
    k_gat_eden<<<g1(E), blk, 0, stream>>>(row, col, a_s, a_d, mf, den, E);
    k_gat_escat<<<edgeW, blk, 0, stream>>>(row, col, a_s, a_d, mf, den, XG, ACC, E);
    k_gat_self_relu<<<g1(NF), blk, 0, stream>>>(a_s, a_d, mf, den, XG, p.gat_b, ACC, X, NF);
    Xin = X;
  }

  // decoder: nmean over outgoing neighbors, neo score, pair MLP
  k_zero_f<<<g1(NF), blk, 0, stream>>>(NSUM, NF);
  k_scat<<<edgeW, blk, 0, stream>>>(col, row, X, NSUM, E);       // NMEAN[row] += z[col]
  k_nmean_div<<<g1(NF), blk, 0, stream>>>(NSUM, deg_out, NF);
  k_neo<<<pairW, blk, 0, stream>>>(psrc, pdst, NSUM, neo_W, neo_b, neo, P);
  k_gather<<<g1(PF), blk, 0, stream>>>(psrc, pdst, X, HU, HV, PF);
  k_gemm128<0,0,0><<<gemmP, blk, 0, stream>>>(HU, decA, nullptr, T1p, P);
  k_gemm128<0,0,1><<<gemmP, blk, 0, stream>>>(HV, decB, nullptr, T1p, P);
  k_dec_final<<<pairW, blk, 0, stream>>>(T1p, neo, dec_W1 + 256 * F, dec_b1,
                                         dec_W2, dec_b2, (float*)d_out, P);
}
